// Model_32409823216440
// MI455X (gfx1250) — matrix operand order repaired and checked by reading, not compiled, not run on hardware
//
#include <hip/hip_runtime.h>
#include <stdint.h>
#include <stddef.h>

typedef __bf16 bf16_t;
typedef __attribute__((ext_vector_type(16))) __bf16 v16bf;
typedef __attribute__((ext_vector_type(8)))  float  v8f;

#define D_FEAT 128
#define MT 4              // M-tiles (16 rows each) per wave

__device__ __forceinline__ bf16_t f2bf(float f) {
  union { float f; uint32_t u; } in; in.f = f;
  uint32_t u = in.u;
  uint32_t r = u + 0x7FFFu + ((u >> 16) & 1u);   // round-to-nearest-even
  union { uint16_t s; bf16_t b; } out; out.s = (uint16_t)(r >> 16);
  return out.b;
}

__global__ void k_zero_f32(float* __restrict__ p, int n) {
  int i = blockIdx.x * blockDim.x + threadIdx.x;
  if (i < n) p[i] = 0.0f;
}

__global__ void k_f32_to_bf16(const float* __restrict__ in, bf16_t* __restrict__ out, int n) {
  int i = blockIdx.x * blockDim.x + threadIdx.x;
  if (i < n) out[i] = f2bf(in[i]);
}

// Wt[n*K + k] = W[k*N + n]   (W is row-major [K][N])
__global__ void k_transpose_bf16(const float* __restrict__ W, bf16_t* __restrict__ Wt,
                                 int K, int N) {
  int i = blockIdx.x * blockDim.x + threadIdx.x;
  if (i >= K * N) return;
  int k = i / N, n = i - k * N;
  Wt[(size_t)n * K + k] = f2bf(W[i]);
}

// One edge per 128 lanes: coalesced row gather + f32 atomic scatter-add.
__global__ void k_scatter_accum(const float* __restrict__ xsrc, const int* __restrict__ src,
                                const int* __restrict__ dst, float* __restrict__ agg,
                                float* __restrict__ cnt, int nedges) {
  int e = blockIdx.x * 2 + (threadIdx.x >> 7);
  int d = threadIdx.x & 127;
  if (e >= nedges) return;
  int s = src[e], t = dst[e];
  atomicAdd(&agg[(size_t)t * D_FEAT + d], xsrc[(size_t)s * D_FEAT + d]);
  if (d == 0) atomicAdd(&cnt[t], 1.0f);
}

__global__ void k_mean_bf16(const float* __restrict__ agg, const float* __restrict__ cnt,
                            bf16_t* __restrict__ out, int total) {
  int i = blockIdx.x * blockDim.x + threadIdx.x;
  if (i >= total) return;
  float c = cnt[i >> 7];
  out[i] = f2bf(agg[i] / fmaxf(c, 1.0f));
}

// out[r,c] = relu?( mean @ Wl + xdst @ Wr + bias )
// 256 threads = 8 waves; block owns rows [r0, r0+64); wave w owns cols [16w, 16w+16)
// across MT=4 row tiles, reusing each B fragment across 4 WMMAs.
// Branchless hot loop: tail tiles clamp their base row (nrows is a multiple of 16),
// only the final stores are guarded.
__global__ __launch_bounds__(256)
void k_sage_gemm(const bf16_t* __restrict__ Amean, const bf16_t* __restrict__ Axdst,
                 const bf16_t* __restrict__ Wlt, const bf16_t* __restrict__ Wrt,
                 const float* __restrict__ bias,
                 float* __restrict__ out_f32, bf16_t* __restrict__ out_bf16,
                 int do_relu, int nrows) {
  const int r0   = blockIdx.x * (16 * MT);
  if (r0 >= nrows) return;                       // uniform across block
  const int wave = threadIdx.x >> 5;
  const int lane = threadIdx.x & 31;
  const int half = lane >> 4;
  const int m    = lane & 15;
  const int col  = wave * 16 + m;

  int tb[MT];                                    // clamped tile base rows (loads in-bounds)
#pragma unroll
  for (int t = 0; t < MT; ++t) {
    int b0 = r0 + 16 * t;
    tb[t] = (b0 < nrows) ? b0 : (nrows - 16);
  }

  v8f c[MT];
  const float bv = bias[col];
#pragma unroll
  for (int t = 0; t < MT; ++t)
#pragma unroll
    for (int r = 0; r < 8; ++r) c[t][r] = bv;

  const bf16_t* bRow1 = Wlt + (size_t)col * D_FEAT;
  const bf16_t* bRow2 = Wrt + (size_t)col * D_FEAT;

#pragma unroll
  for (int k0 = 0; k0 < D_FEAT; k0 += 32) {
    v16bf b;
    const bf16_t* q = bRow1 + k0 + half * 8;     // B: K = k0 + 16*(e>>3) + 8*half + (e&7)
#pragma unroll
    for (int e = 0; e < 8; ++e) { b[e] = q[e]; b[8 + e] = q[16 + e]; }
#pragma unroll
    for (int t = 0; t < MT; ++t) {
      v16bf a;                                   // A: K = k0 + 16*(e>>3) + 8*half + (e&7)
      const bf16_t* p = Amean + (size_t)(tb[t] + m) * D_FEAT + k0 + half * 8;
#pragma unroll
      for (int e = 0; e < 8; ++e) { a[e] = p[e]; a[8 + e] = p[16 + e]; }
      c[t] = __builtin_amdgcn_wmma_f32_16x16x32_bf16(false, a, false, b, (short)0,
                                                     c[t], false, false);
    }
  }
#pragma unroll
  for (int k0 = 0; k0 < D_FEAT; k0 += 32) {
    v16bf b;
    const bf16_t* q = bRow2 + k0 + half * 8;
#pragma unroll
    for (int e = 0; e < 8; ++e) { b[e] = q[e]; b[8 + e] = q[16 + e]; }
#pragma unroll
    for (int t = 0; t < MT; ++t) {
      v16bf a;
      const bf16_t* p = Axdst + (size_t)(tb[t] + m) * D_FEAT + k0 + half * 8;
#pragma unroll
      for (int e = 0; e < 8; ++e) { a[e] = p[e]; a[8 + e] = p[16 + e]; }
      c[t] = __builtin_amdgcn_wmma_f32_16x16x32_bf16(false, a, false, b, (short)0,
                                                     c[t], false, false);
    }
  }

#pragma unroll
  for (int t = 0; t < MT; ++t) {
    if (r0 + 16 * t < nrows) {                   // uniform store guard (once, not in k-loop)
#pragma unroll
      for (int r = 0; r < 8; ++r) {              // C: M = r + 8*half, N = m
        int row = r0 + 16 * t + half * 8 + r;
        float v = c[t][r];
        if (do_relu) v = fmaxf(v, 0.0f);
        size_t idx = (size_t)row * D_FEAT + col;
        if (out_f32)  out_f32[idx]  = v;
        if (out_bf16) out_bf16[idx] = f2bf(v);
      }
    }
  }
}

// decoder: h = relu(concat(zdrug[row], zdis[col]) @ dec_W1 + b1); out = h @ dec_W2 + b2
// Block owns 64 label-edges; wave owns a 16-col slab across MT=4 row tiles.
__global__ __launch_bounds__(256)
void k_decoder(const bf16_t* __restrict__ zdrug, const bf16_t* __restrict__ zdis,
               const int* __restrict__ rowI, const int* __restrict__ colI,
               const bf16_t* __restrict__ W1t, const float* __restrict__ b1,
               const float* __restrict__ W2, const float* __restrict__ b2,
               float* __restrict__ out, int L) {
  __shared__ float hs[16 * MT * D_FEAT];         // 32 KB
  const int r0   = blockIdx.x * (16 * MT);
  if (r0 >= L) return;                           // uniform across block
  const int wave = threadIdx.x >> 5;
  const int lane = threadIdx.x & 31;
  const int half = lane >> 4;
  const int m    = lane & 15;
  const int col  = wave * 16 + m;

  const bf16_t* aL[MT];
  const bf16_t* aR[MT];
#pragma unroll
  for (int t = 0; t < MT; ++t) {
    int b0 = r0 + 16 * t;
    int tbase = ((b0 < L) ? b0 : (L - 16)) + m;  // clamped: gathers stay in-bounds
    aL[t] = zdrug + (size_t)rowI[tbase] * D_FEAT;
    aR[t] = zdis  + (size_t)colI[tbase] * D_FEAT;
  }
  const bf16_t* bR = W1t + (size_t)col * 256;

  v8f c[MT];
  const float bv = b1[col];
#pragma unroll
  for (int t = 0; t < MT; ++t)
#pragma unroll
    for (int r = 0; r < 8; ++r) c[t][r] = bv;

#pragma unroll
  for (int k0 = 0; k0 < 256; k0 += 32) {
    v16bf b;
    const bf16_t* q = bR + k0 + half * 8;
#pragma unroll
    for (int e = 0; e < 8; ++e) { b[e] = q[e]; b[8 + e] = q[16 + e]; }
#pragma unroll
    for (int t = 0; t < MT; ++t) {
      v16bf a;
      const bf16_t* p = (k0 < 128) ? (aL[t] + k0 + half * 8)
                                   : (aR[t] + (k0 - 128) + half * 8);
#pragma unroll
      for (int e = 0; e < 8; ++e) { a[e] = p[e]; a[8 + e] = p[16 + e]; }
      c[t] = __builtin_amdgcn_wmma_f32_16x16x32_bf16(false, a, false, b, (short)0,
                                                     c[t], false, false);
    }
  }

#pragma unroll
  for (int t = 0; t < MT; ++t)                   // LDS writes always safe
#pragma unroll
    for (int r = 0; r < 8; ++r)
      hs[(16 * t + half * 8 + r) * D_FEAT + col] = fmaxf(c[t][r], 0.0f);
  __syncthreads();

  if (threadIdx.x < 16 * MT && r0 + (int)threadIdx.x < L) {
    float acc = b2[0];
    const float* hp = &hs[threadIdx.x * D_FEAT];
#pragma unroll 8
    for (int k = 0; k < D_FEAT; ++k) acc += hp[k] * W2[k];
    out[r0 + threadIdx.x] = acc;
  }
}

static inline int cdiv(int a, int b) { return (a + b - 1) / b; }

extern "C" void kernel_launch(void* const* d_in, const int* in_sizes, int n_in,
                              void* d_out, int out_size, void* d_ws, size_t ws_size,
                              hipStream_t stream) {
  (void)n_in; (void)ws_size; (void)out_size;
  const float* x_drug = (const float*)d_in[0];
  const float* x_dis  = (const float*)d_in[1];
  const int*   ei_d2s = (const int*)d_in[2];
  const int*   ei_s2d = (const int*)d_in[3];
  const int*   eli    = (const int*)d_in[4];
  const float* Wmat[8];
  for (int i = 0; i < 8; ++i) Wmat[i] = (const float*)d_in[5 + i];
  const float* b1_d2s = (const float*)d_in[13];
  const float* b1_s2d = (const float*)d_in[14];
  const float* b2_d2s = (const float*)d_in[15];
  const float* b2_s2d = (const float*)d_in[16];
  const float* dec_W1 = (const float*)d_in[17];
  const float* dec_b1 = (const float*)d_in[18];
  const float* dec_W2 = (const float*)d_in[19];
  const float* dec_b2 = (const float*)d_in[20];

  const int N = in_sizes[0] / D_FEAT;            // 20000
  const int E = in_sizes[2] / 2;                 // 640000
  const int L = in_sizes[4] / 2;                 // 200000
  const int ND = N * D_FEAT;

  // ---- workspace carve-out ----
  char* ws = (char*)d_ws;
  size_t off = 0;
  auto carve = [&](size_t bytes) -> char* {
    char* p = ws + off;
    off += (bytes + 255) & ~(size_t)255;
    return p;
  };
  float*  agg      = (float*)carve((size_t)ND * 4);
  float*  cnt      = (float*)carve((size_t)N * 4);
  bf16_t* mean_bf  = (bf16_t*)carve((size_t)ND * 2);
  bf16_t* xdrug_bf = (bf16_t*)carve((size_t)ND * 2);
  bf16_t* xdis_bf  = (bf16_t*)carve((size_t)ND * 2);
  float*  hdrug_f  = (float*)carve((size_t)ND * 4);
  float*  hdis_f   = (float*)carve((size_t)ND * 4);
  bf16_t* hdrug_bf = (bf16_t*)carve((size_t)ND * 2);
  bf16_t* hdis_bf  = (bf16_t*)carve((size_t)ND * 2);
  bf16_t* zdrug_bf = (bf16_t*)carve((size_t)ND * 2);
  bf16_t* zdis_bf  = (bf16_t*)carve((size_t)ND * 2);
  bf16_t* Wt[8];
  for (int i = 0; i < 8; ++i) Wt[i] = (bf16_t*)carve((size_t)D_FEAT * D_FEAT * 2);
  bf16_t* decW1t = (bf16_t*)carve((size_t)256 * D_FEAT * 2);

  const int TB = 256;
  for (int i = 0; i < 8; ++i)
    k_transpose_bf16<<<cdiv(D_FEAT * D_FEAT, TB), TB, 0, stream>>>(Wmat[i], Wt[i], D_FEAT, D_FEAT);
  k_transpose_bf16<<<cdiv(256 * D_FEAT, TB), TB, 0, stream>>>(dec_W1, decW1t, 256, D_FEAT);
  k_f32_to_bf16<<<cdiv(ND, TB), TB, 0, stream>>>(x_drug, xdrug_bf, ND);
  k_f32_to_bf16<<<cdiv(ND, TB), TB, 0, stream>>>(x_dis,  xdis_bf,  ND);

  const int gemm_blocks = cdiv(N, 16 * MT);
  const int scat_blocks = cdiv(E, 2);

  // ---- layer 1, d2s (dst = disease)
  k_zero_f32<<<cdiv(ND, TB), TB, 0, stream>>>(agg, ND);
  k_zero_f32<<<cdiv(N, TB), TB, 0, stream>>>(cnt, N);
  k_scatter_accum<<<scat_blocks, TB, 0, stream>>>(x_drug, ei_d2s, ei_d2s + E, agg, cnt, E);
  k_mean_bf16<<<cdiv(ND, TB), TB, 0, stream>>>(agg, cnt, mean_bf, ND);
  k_sage_gemm<<<gemm_blocks, TB, 0, stream>>>(mean_bf, xdis_bf, Wt[0], Wt[1], b1_d2s,
                                              hdis_f, hdis_bf, 1, N);
  // ---- layer 1, s2d (dst = drug)
  k_zero_f32<<<cdiv(ND, TB), TB, 0, stream>>>(agg, ND);
  k_zero_f32<<<cdiv(N, TB), TB, 0, stream>>>(cnt, N);
  k_scatter_accum<<<scat_blocks, TB, 0, stream>>>(x_dis, ei_s2d, ei_s2d + E, agg, cnt, E);
  k_mean_bf16<<<cdiv(ND, TB), TB, 0, stream>>>(agg, cnt, mean_bf, ND);
  k_sage_gemm<<<gemm_blocks, TB, 0, stream>>>(mean_bf, xdrug_bf, Wt[2], Wt[3], b1_s2d,
                                              hdrug_f, hdrug_bf, 1, N);
  // ---- layer 2, d2s: z_dis (no relu)
  k_zero_f32<<<cdiv(ND, TB), TB, 0, stream>>>(agg, ND);
  k_zero_f32<<<cdiv(N, TB), TB, 0, stream>>>(cnt, N);
  k_scatter_accum<<<scat_blocks, TB, 0, stream>>>(hdrug_f, ei_d2s, ei_d2s + E, agg, cnt, E);
  k_mean_bf16<<<cdiv(ND, TB), TB, 0, stream>>>(agg, cnt, mean_bf, ND);
  k_sage_gemm<<<gemm_blocks, TB, 0, stream>>>(mean_bf, hdis_bf, Wt[4], Wt[5], b2_d2s,
                                              (float*)nullptr, zdis_bf, 0, N);
  // ---- layer 2, s2d: z_drug (no relu)
  k_zero_f32<<<cdiv(ND, TB), TB, 0, stream>>>(agg, ND);
  k_zero_f32<<<cdiv(N, TB), TB, 0, stream>>>(cnt, N);
  k_scatter_accum<<<scat_blocks, TB, 0, stream>>>(hdis_f, ei_s2d, ei_s2d + E, agg, cnt, E);
  k_mean_bf16<<<cdiv(ND, TB), TB, 0, stream>>>(agg, cnt, mean_bf, ND);
  k_sage_gemm<<<gemm_blocks, TB, 0, stream>>>(mean_bf, hdrug_bf, Wt[6], Wt[7], b2_s2d,
                                              (float*)nullptr, zdrug_bf, 0, N);
  // ---- decoder
  k_decoder<<<cdiv(L, 16 * MT), TB, 0, stream>>>(zdrug_bf, zdis_bf, eli, eli + L,
                                                 decW1t, dec_b1, dec_W2, dec_b2,
                                                 (float*)d_out, L);
}
